// RvNN_71829033058694
// MI455X (gfx1250) — matrix operand order repaired and checked by reading, not compiled, not run on hardware
//
#include <hip/hip_runtime.h>
#include <math.h>

typedef __attribute__((ext_vector_type(16))) _Float16 v16h;
typedef __attribute__((ext_vector_type(8)))  float    v8f;

#define HDIM 64
#define LDIM 512

// ---------------------------------------------------------------------------
// Kernel 0: zero node_h
// ---------------------------------------------------------------------------
__global__ void k_zero(float* __restrict__ p, int n) {
    int i = blockIdx.x * blockDim.x + threadIdx.x;
    if (i < n) p[i] = 0.0f;
}

// ---------------------------------------------------------------------------
// Kernel 1: xe[node][:] = sum_l x_word[node][l] * embedding[x_index[node][l]][:]
// One block (256 threads) per node; 4 segments x 64 columns, LDS reduction.
// ---------------------------------------------------------------------------
__global__ void k_embed(const float* __restrict__ xw, const int* __restrict__ xi,
                        const float* __restrict__ emb, float* __restrict__ xe) {
    __shared__ float red[256];
    int node = blockIdx.x;
    int col  = threadIdx.x & 63;
    int seg  = threadIdx.x >> 6;           // 0..3
    const float* wrow = xw + (size_t)node * LDIM;
    const int*   irow = xi + (size_t)node * LDIM;
    float acc = 0.0f;
    for (int l = seg; l < LDIM; l += 4) {
        float w  = wrow[l];                // broadcast across 64 lanes
        int   id = irow[l];
        acc += w * emb[(size_t)id * HDIM + col];   // coalesced 256B row
    }
    red[threadIdx.x] = acc;
    __syncthreads();
    if (seg == 0) {
        acc = red[col] + red[col + 64] + red[col + 128] + red[col + 192];
        xe[(size_t)node * HDIM + col] = acc;
    }
}

// ---------------------------------------------------------------------------
// Kernel 2: A_m[i][:] = W_m @ xe[i] + b_m  for m in {z, r, h}
// GEMM  (n_nodes x 64) @ (64 x 64)^T  via v_wmma_f32_16x16x32_f16.
// One wave (32 threads) per 16x16 output tile. grid = (n_nodes/16, 12):
//   blockIdx.y: [0..3]=W_z cols, [4..7]=W_r cols, [8..11]=W_h cols.
// Y[i][n] = sum_k xe[i][k] * W[n][k]  ->  A = xe (MxK), B[k][n] = W[n][k].
// ---------------------------------------------------------------------------
__global__ void k_gemm(const float* __restrict__ xe,
                       const float* __restrict__ Wz, const float* __restrict__ Wr,
                       const float* __restrict__ Wh,
                       const float* __restrict__ bz, const float* __restrict__ br,
                       const float* __restrict__ bh,
                       float* __restrict__ Az, float* __restrict__ Ar,
                       float* __restrict__ Ah) {
    int lane  = threadIdx.x;               // 0..31 (wave32, EXEC all ones)
    int tileM = blockIdx.x;
    int y     = blockIdx.y;
    int mat   = y >> 2;
    int ct    = y & 3;                     // 16-column tile within the 64 cols

    const float* W; const float* bias; float* out;
    if (mat == 0)      { W = Wz; bias = bz; out = Az; }
    else if (mat == 1) { W = Wr; bias = br; out = Ar; }
    else               { W = Wh; bias = bh; out = Ah; }

    int half = lane >> 4;                  // 0 | 1
    int mn   = lane & 15;                  // row (A) / col (B) index

    // A fragment rows: lane holds row (tileM*16 + mn) of xe
    const float* arow = xe + ((size_t)tileM * 16 + mn) * HDIM;
    // B fragment cols: B[k][n] = W[n][k] -> lane holds row (ct*16 + mn) of W
    const float* brow = W + ((size_t)ct * 16 + mn) * HDIM;

    int kbA = half * 8;                    // A: lanes16-31 offset K by +8
    int kbB = half * 16;                   // B: lanes16-31 hold K=8..15, 24..31 of block

    v8f c = {};
    #pragma unroll
    for (int kblk = 0; kblk < HDIM; kblk += 32) {
        v16h a, b;
        #pragma unroll
        for (int t = 0; t < 8; ++t) {
            // A 16x32 f16 layout: VGPR0-3 -> K = kbA+0..7, VGPR4-7 -> K = kbA+16..23
            a[t]     = (_Float16)arow[kblk + kbA + t];
            a[t + 8] = (_Float16)arow[kblk + kbA + 16 + t];
        }
        #pragma unroll
        for (int t = 0; t < 16; ++t) {
            // B 32x16 f16 layout: matches A: t<8 -> K = 8*half+t, t>=8 -> K = 8*half+16+(t-8)
            b[t] = (_Float16)brow[kblk + (kbB >> 1) + (t & 7) + ((t >> 3) << 4)];
        }
        c = __builtin_amdgcn_wmma_f32_16x16x32_f16(
                /*neg_a=*/false, a, /*neg_b=*/false, b,
                /*c_mod=*/(short)0, c, /*reuse_a=*/false, /*reuse_b=*/false);
    }

    // D layout: VGPR j -> M = j + 8*half (within tile), N = lane&15
    float bv = bias[ct * 16 + mn];
    #pragma unroll
    for (int j = 0; j < 8; ++j) {
        int row = tileM * 16 + j + 8 * half;
        out[(size_t)row * HDIM + ct * 16 + mn] = c[j] + bv;
    }
}

// ---------------------------------------------------------------------------
// Kernel 3: sequential tree-GRU scan. Single block, 64 threads (2 waves).
// Thread j owns hidden column j for the entire scan (same-lane global
// store->load recurrence on node_h => hardware-ordered, no fence needed).
// U matrices live in LDS with stride-65 padding (conflict-free matvec).
// ---------------------------------------------------------------------------
__device__ __forceinline__ float hsig(float x) {
    return fminf(fmaxf(0.2f * x + 0.5f, 0.0f), 1.0f);
}

__global__ void k_scan(const int* __restrict__ tree, const int* __restrict__ npp,
                       const float* __restrict__ Uz, const float* __restrict__ Ur,
                       const float* __restrict__ Uh,
                       const float* __restrict__ Az, const float* __restrict__ Ar,
                       const float* __restrict__ Ah,
                       float* __restrict__ node_h, float* __restrict__ out,
                       int n_nodes) {
    __shared__ float sUz[HDIM * 65];
    __shared__ float sUr[HDIM * 65];
    __shared__ float sUh[HDIM * 65];
    __shared__ float phs[HDIM];
    __shared__ float prv[HDIM];

    int j = threadIdx.x;                   // 0..63
    for (int t = j; t < HDIM * HDIM; t += 64) {
        int rr = t >> 6, cc = t & 63;
        sUz[rr * 65 + cc] = Uz[t];
        sUr[rr * 65 + cc] = Ur[t];
        sUh[rr * 65 + cc] = Uh[t];
    }
    __syncthreads();

    int   np   = npp[0];
    float vmax = -1e30f;

    for (int i = 0; i < n_nodes - 1; ++i) {
        int p  = tree[2 * i];
        int ci = tree[2 * i + 1];

        phs[j] = node_h[(size_t)p * HDIM + j];   // column j written by thread j
        __syncthreads();

        float az = Az[(size_t)i * HDIM + j];
        float ar = Ar[(size_t)i * HDIM + j];
        float ah = Ah[(size_t)i * HDIM + j];

        float uz = 0.0f, ur = 0.0f;
        #pragma unroll 8
        for (int k = 0; k < HDIM; ++k) {
            float ph = phs[k];                   // LDS broadcast
            uz += sUz[j * 65 + k] * ph;          // bank = (j+k)&63: conflict-free
            ur += sUr[j * 65 + k] * ph;
        }
        float z = hsig(az + uz);
        float r = hsig(ar + ur);

        prv[j] = phs[j] * r;
        __syncthreads();

        float uh = 0.0f;
        #pragma unroll 8
        for (int k = 0; k < HDIM; ++k)
            uh += sUh[j * 65 + k] * prv[k];

        float cg = tanhf(ah + uh);
        float h  = z * phs[j] + (1.0f - z) * cg;

        node_h[(size_t)ci * HDIM + j] = h;
        if (i >= np - 1) vmax = fmaxf(vmax, h);
        __syncthreads();                         // protect phs/prv for next iter
    }
    out[j] = vmax;
}

// ---------------------------------------------------------------------------
// Launcher
// Inputs: 0 x_word, 1 x_index, 2 tree, 3 num_parent, 4 embedding,
//         5 W_z, 6 U_z, 7 b_z, 8 W_r, 9 U_r, 10 b_r, 11 W_h, 12 U_h, 13 b_h
// Workspace: xe | Az | Ar | Ah | node_h  (5 * n_nodes * 64 floats = 10 MB)
// ---------------------------------------------------------------------------
extern "C" void kernel_launch(void* const* d_in, const int* in_sizes, int n_in,
                              void* d_out, int out_size, void* d_ws, size_t ws_size,
                              hipStream_t stream) {
    const float* x_word  = (const float*)d_in[0];
    const int*   x_index = (const int*)  d_in[1];
    const int*   tree    = (const int*)  d_in[2];
    const int*   np      = (const int*)  d_in[3];
    const float* emb     = (const float*)d_in[4];
    const float* Wz      = (const float*)d_in[5];
    const float* Uz      = (const float*)d_in[6];
    const float* bz      = (const float*)d_in[7];
    const float* Wr      = (const float*)d_in[8];
    const float* Ur      = (const float*)d_in[9];
    const float* br      = (const float*)d_in[10];
    const float* Wh      = (const float*)d_in[11];
    const float* Uh      = (const float*)d_in[12];
    const float* bh      = (const float*)d_in[13];
    float* out = (float*)d_out;

    int n_nodes = in_sizes[0] / LDIM;          // 8192
    size_t mat = (size_t)n_nodes * HDIM;

    float* xe = (float*)d_ws;
    float* Az = xe + mat;
    float* Ar = Az + mat;
    float* Ah = Ar + mat;
    float* nh = Ah + mat;

    k_zero<<<(int)((mat + 255) / 256), 256, 0, stream>>>(nh, (int)mat);
    k_embed<<<n_nodes, 256, 0, stream>>>(x_word, x_index, emb, xe);

    dim3 g2(n_nodes / 16, 12);
    k_gemm<<<g2, 32, 0, stream>>>(xe, Wz, Wr, Wh, bz, br, bh, Az, Ar, Ah);

    k_scan<<<1, 64, 0, stream>>>(tree, np, Uz, Ur, Uh, Az, Ar, Ah, nh, out, n_nodes);
}
